// OptimizedParallelHierarchicalAttention_73495480369477
// MI455X (gfx1250) — matrix operand order repaired and checked by reading, not compiled, not run on hardware
//
#include <hip/hip_runtime.h>
#include <hip/hip_bf16.h>

typedef __bf16 v8bf  __attribute__((ext_vector_type(8)));
typedef __bf16 v16bf __attribute__((ext_vector_type(16)));
typedef float  v8f   __attribute__((ext_vector_type(8)));

#define DD   1024
#define HH   16
#define DHD  64
#define BSZ  256
#define NWG  64        // B*NB = 4*16 blocks
#define MTOK 16384     // B*S

// ---- WMMA fragment helpers (16-bit operands, contiguous k order used by this file, wave32) ----------------
// A (16x32): lane l<16 -> row l, K=0..15 contiguous; lanes 16..31 -> K=16..31
__device__ inline v16bf ld_a(const __bf16* p, int lh) {
  return *(const v16bf*)(p + lh * 16);
}
// B (32x16): lane l<16 -> col l, K=0..15 contiguous; lanes 16..31 -> K=16..31
__device__ inline v16bf ld_b(const __bf16* p) { return *(const v16bf*)p; }

__device__ inline v8f wmma_bf16(v16bf a, v16bf b, v8f c) {
  return __builtin_amdgcn_wmma_f32_16x16x32_bf16(false, a, false, b, (short)0, c,
                                                 false, false);
}

// ---- CDNA5 async global->LDS DMA (ASYNCcnt-tracked) ----------------------
// Generic pointer to shared memory: low 32 bits are the LDS byte address
// (aperture layout, ISA 10.2).
__device__ inline unsigned lds_addr(const void* p) { return (unsigned)(size_t)p; }

// Copy 32 bytes global->LDS: INST_OFFSET applies to both LDS and global
// addresses, so one base pair covers both halves.
__device__ inline void async_copy32(unsigned ldsoff, const void* gptr) {
  asm volatile("global_load_async_to_lds_b128 %0, %1, off\n\t"
               "global_load_async_to_lds_b128 %0, %1, off offset:16"
               :: "v"(ldsoff), "v"(gptr) : "memory");
}
__device__ inline void wait_async0() {
  asm volatile("s_wait_asynccnt 0x0" ::: "memory");
}

// ---- conversions ---------------------------------------------------------
__global__ void cvt_bf16(const float* __restrict__ in, __bf16* __restrict__ out, int n) {
  int i = blockIdx.x * 256 + threadIdx.x;
  if (i < n) out[i] = (__bf16)in[i];
}
// 1024x1024 weight: out[n][k] = in[k][n]  (transposed bf16 copy)
__global__ void cvt_t_bf16(const float* __restrict__ in, __bf16* __restrict__ out) {
  int i = blockIdx.x * 256 + threadIdx.x;       // 0..1M-1
  int n = i >> 10, k = i & 1023;
  out[i] = (__bf16)in[k * DD + n];
}

// ---- tiled WMMA GEMM: C[M,1024] = A[M,1024] @ Wt^T + bias ----------------
// Wt is stored transposed [n][k]. outf!=null -> f32 out else bf16 out.
// addbuf!=null -> += addbuf[(row>>8)*1024 + col] (block-summary broadcast).
// Double-buffered LDS staged via async global->LDS DMA.
__global__ __launch_bounds__(256) void gemm_wmma(
    const __bf16* __restrict__ A, const __bf16* __restrict__ Wt,
    const float* __restrict__ bias, __bf16* __restrict__ outb,
    float* __restrict__ outf, const float* __restrict__ addbuf, int M)
{
  __shared__ __bf16 As[2][128][32];
  __shared__ __bf16 Bs[2][128][32];
  const int tid = threadIdx.x;
  const int wave = tid >> 5, ln = tid & 31;
  const int l16 = ln & 15, lh = ln >> 4;
  const int wm = wave >> 1, wn = wave & 1;   // 4x2 wave grid -> 32x64 per wave
  const int tm = blockIdx.y * 128;
  const int tn = blockIdx.x * 128;

  // staging assignment: each thread owns one 32B segment per matrix per tile
  const int lr = tid >> 1, lhalf = tid & 1;
  int grow = tm + lr; if (grow >= M) grow = M - 1;
  const __bf16* aptr = A  + (size_t)grow * DD + lhalf * 16;
  const __bf16* bptr = Wt + (size_t)(tn + lr) * DD + lhalf * 16;
  const unsigned aoff0 = lds_addr(&As[0][lr][lhalf * 16]);
  const unsigned aoff1 = lds_addr(&As[1][lr][lhalf * 16]);
  const unsigned boff0 = lds_addr(&Bs[0][lr][lhalf * 16]);
  const unsigned boff1 = lds_addr(&Bs[1][lr][lhalf * 16]);

  // prologue: DMA tile 0 into buffer 0
  async_copy32(aoff0, aptr);
  async_copy32(boff0, bptr);

  v8f acc[2][4] = {};
  for (int kt = 0; kt < 32; ++kt) {
    const int cur = kt & 1;
    wait_async0();       // this wave's pending tile DMA is in LDS
    __syncthreads();     // all waves' DMA visible; prev buffer fully consumed
    if (kt + 1 < 32) {   // DMA next tile into the other buffer, overlap compute
      async_copy32(cur ? aoff0 : aoff1, aptr + (kt + 1) * 32);
      async_copy32(cur ? boff0 : boff1, bptr + (kt + 1) * 32);
    }
#pragma unroll
    for (int i = 0; i < 2; ++i) {
      v16bf a = ld_a(&As[cur][wm * 32 + i * 16 + l16][0], lh);
#pragma unroll
      for (int j = 0; j < 4; ++j) {
        v16bf b = ld_b(&Bs[cur][wn * 64 + j * 16 + l16][lh * 16]);
        acc[i][j] = wmma_bf16(a, b, acc[i][j]);
      }
    }
  }
#pragma unroll
  for (int i = 0; i < 2; ++i)
#pragma unroll
    for (int j = 0; j < 4; ++j)
#pragma unroll
      for (int r = 0; r < 8; ++r) {
        const int row = tm + wm * 32 + i * 16 + r + lh * 8;
        const int col = tn + wn * 64 + j * 16 + l16;
        if (row < M) {
          float v = acc[i][j][r] + bias[col];
          if (addbuf) v += addbuf[(size_t)(row >> 8) * DD + col];
          if (outf) outf[(size_t)row * DD + col] = v;
          else      outb[(size_t)row * DD + col] = (__bf16)v;
        }
      }
}

// ---- local block attention, flash style, one WG per (block, head) --------
__global__ __launch_bounds__(256) void local_attn(
    const __bf16* __restrict__ Q, const __bf16* __restrict__ K,
    const __bf16* __restrict__ V, const float* __restrict__ bias,
    __bf16* __restrict__ ctx)
{
  __shared__ __bf16 Vt[DHD][BSZ];       // V transposed, 32 KB
  __shared__ __bf16 Pbuf[8][32][64];    // per-wave P re-layout, 32 KB
  const int blk = blockIdx.x;           // 0..63
  const int h   = blockIdx.y;           // 0..15
  const size_t tok0 = (size_t)blk * BSZ;
  const int hb = h * DHD;
  const int tid = threadIdx.x;
  const int wave = tid >> 5, ln = tid & 31;
  const int l16 = ln & 15, lh = ln >> 4;

  { // stage V^T into LDS
    const __bf16* vp = V + (tok0 + tid) * DD + hb;
#pragma unroll
    for (int d = 0; d < DHD; ++d) Vt[d][tid] = vp[d];
  }
  __syncthreads();

  const int m0 = wave * 32;   // 32 query rows per wave
  v16bf qa[2][2];
#pragma unroll
  for (int i = 0; i < 2; ++i)
#pragma unroll
    for (int kk = 0; kk < 2; ++kk)
      qa[i][kk] = ld_a(Q + (tok0 + m0 + i * 16 + l16) * DD + hb + kk * 32, lh);

  v8f acc[2][4] = {};
  float rmax[2][8], rsum[2][8];
#pragma unroll
  for (int i = 0; i < 2; ++i)
#pragma unroll
    for (int r = 0; r < 8; ++r) { rmax[i][r] = -1e30f; rsum[i][r] = 0.f; }

  const float* bp = bias + (size_t)h * BSZ * BSZ;

  for (int nc = 0; nc < 4; ++nc) {          // key chunks of 64
    const int nb = nc * 64;
    v8f s[2][4] = {};
#pragma unroll
    for (int kk = 0; kk < 2; ++kk) {
      v16bf kb[4];
#pragma unroll
      for (int j = 0; j < 4; ++j)
        kb[j] = ld_b(K + (tok0 + nb + j * 16 + l16) * DD + hb + kk * 32 + lh * 16);
#pragma unroll
      for (int i = 0; i < 2; ++i)
#pragma unroll
        for (int j = 0; j < 4; ++j)
          s[i][j] = wmma_bf16(qa[i][kk], kb[j], s[i][j]);
    }
    // scale + position bias + online softmax (rows split across lane halves)
#pragma unroll
    for (int i = 0; i < 2; ++i) {
#pragma unroll
      for (int r = 0; r < 8; ++r) {
        const int row = m0 + i * 16 + r + lh * 8;
        float mx = -1e30f;
#pragma unroll
        for (int j = 0; j < 4; ++j) {
          const int col = nb + j * 16 + l16;
          float v = s[i][j][r] * 0.125f + bp[row * BSZ + col];
          s[i][j][r] = v;
          mx = fmaxf(mx, v);
        }
#pragma unroll
        for (int o = 1; o < 16; o <<= 1) mx = fmaxf(mx, __shfl_xor(mx, o, 16));
        const float nm = fmaxf(rmax[i][r], mx);
        const float sc = __expf(rmax[i][r] - nm);
        rmax[i][r] = nm;
        rsum[i][r] *= sc;
#pragma unroll
        for (int j = 0; j < 4; ++j) acc[i][j][r] *= sc;
        float ps = 0.f;
#pragma unroll
        for (int j = 0; j < 4; ++j) {
          float p = __expf(s[i][j][r] - nm);
          s[i][j][r] = p;
          ps += p;
        }
#pragma unroll
        for (int o = 1; o < 16; o <<= 1) ps += __shfl_xor(ps, o, 16);
        rsum[i][r] += ps;
      }
    }
    // P (C-layout) -> per-wave LDS -> A-fragments, then ctx += P @ V
#pragma unroll
    for (int i = 0; i < 2; ++i)
#pragma unroll
      for (int j = 0; j < 4; ++j)
#pragma unroll
        for (int r = 0; r < 8; ++r)
          Pbuf[wave][i * 16 + r + lh * 8][j * 16 + l16] = (__bf16)s[i][j][r];
    __syncthreads();
#pragma unroll
    for (int kk = 0; kk < 2; ++kk) {
      v16bf pa[2];
#pragma unroll
      for (int i = 0; i < 2; ++i)
        pa[i] = ld_a(&Pbuf[wave][i * 16 + l16][kk * 32], lh);
      v16bf vb[4];
#pragma unroll
      for (int j = 0; j < 4; ++j)
        vb[j] = ld_b(&Vt[j * 16 + l16][nb + kk * 32 + lh * 16]);
#pragma unroll
      for (int i = 0; i < 2; ++i)
#pragma unroll
        for (int j = 0; j < 4; ++j)
          acc[i][j] = wmma_bf16(pa[i], vb[j], acc[i][j]);
    }
    __syncthreads();
  }
  // normalize and store bf16 ctx
#pragma unroll
  for (int i = 0; i < 2; ++i)
#pragma unroll
    for (int j = 0; j < 4; ++j)
#pragma unroll
      for (int r = 0; r < 8; ++r) {
        const int row = m0 + i * 16 + r + lh * 8;
        const int col = j * 16 + l16;
        float v = acc[i][j][r] / rsum[i][r];
        ctx[(tok0 + row) * DD + hb + col] = (__bf16)v;
      }
}

// ---- block mean-pool: mp[g][d] = mean over 256 tokens of x ---------------
__global__ __launch_bounds__(256) void meanpool(const float* __restrict__ x,
                                                __bf16* __restrict__ mp) {
  const int g = blockIdx.x;                 // 0..63 (b*16+blk)
  const size_t base = (size_t)g * BSZ * DD;
  const int tid = threadIdx.x;
#pragma unroll
  for (int c = 0; c < 4; ++c) {
    const int d = tid + c * 256;
    float s = 0.f;
    for (int t = 0; t < BSZ; ++t) s += x[base + (size_t)t * DD + d];
    mp[(size_t)g * DD + d] = (__bf16)(s * (1.0f / BSZ));
  }
}

// ---- tiny 16-token summary attention, one WG per (b, h) ------------------
__global__ __launch_bounds__(256) void summ_attn(
    const __bf16* __restrict__ Q, const __bf16* __restrict__ K,
    const __bf16* __restrict__ V, __bf16* __restrict__ ctx)
{
  __shared__ float S[16][17];
  __shared__ float P[16][17];
  const int b = blockIdx.x, h = blockIdx.y;
  const int hb = h * DHD;
  const int tid = threadIdx.x;
  const int q = tid >> 4, kk = tid & 15;
  {
    const __bf16* qp = Q + ((size_t)b * 16 + q) * DD + hb;
    const __bf16* kp = K + ((size_t)b * 16 + kk) * DD + hb;
    float a = 0.f;
#pragma unroll
    for (int d = 0; d < DHD; ++d) a += (float)qp[d] * (float)kp[d];
    S[q][kk] = a * 0.125f;
  }
  __syncthreads();
  if (tid < 16) {
    float mx = -1e30f;
    for (int c = 0; c < 16; ++c) mx = fmaxf(mx, S[tid][c]);
    float sum = 0.f;
    for (int c = 0; c < 16; ++c) { float p = __expf(S[tid][c] - mx); P[tid][c] = p; sum += p; }
    const float inv = 1.f / sum;
    for (int c = 0; c < 16; ++c) P[tid][c] *= inv;
  }
  __syncthreads();
#pragma unroll
  for (int e = 0; e < 4; ++e) {
    const int idx = tid + e * 256;          // 16 rows x 64 cols
    const int r = idx >> 6, d = idx & 63;
    float a = 0.f;
    for (int c = 0; c < 16; ++c)
      a += P[r][c] * (float)V[((size_t)b * 16 + c) * DD + hb + d];
    ctx[((size_t)b * 16 + r) * DD + hb + d] = (__bf16)a;
  }
}

// ---- host-side orchestration --------------------------------------------
extern "C" void kernel_launch(void* const* d_in, const int* in_sizes, int n_in,
                              void* d_out, int out_size, void* d_ws, size_t ws_size,
                              hipStream_t stream) {
  (void)in_sizes; (void)n_in; (void)out_size; (void)ws_size;
  const float* x        = (const float*)d_in[0];
  const float* w_f32[9] = { (const float*)d_in[1],  (const float*)d_in[3],
                            (const float*)d_in[5],  (const float*)d_in[7],
                            (const float*)d_in[10], (const float*)d_in[12],
                            (const float*)d_in[14], (const float*)d_in[16],
                            (const float*)d_in[18] };           // lq lk lv lo sq sk sv so sum
  const float* lq_b = (const float*)d_in[2];
  const float* lk_b = (const float*)d_in[4];
  const float* lv_b = (const float*)d_in[6];
  const float* lo_b = (const float*)d_in[8];
  const float* pos_bias = (const float*)d_in[9];
  const float* sq_b = (const float*)d_in[11];
  const float* sk_b = (const float*)d_in[13];
  const float* sv_b = (const float*)d_in[15];
  const float* so_b = (const float*)d_in[17];
  const float* sum_b = (const float*)d_in[19];

  char* ws = (char*)d_ws;
  size_t off = 0;
  auto alloc = [&](size_t bytes) -> void* {
    void* p = ws + off;
    off += (bytes + 255) & ~(size_t)255;
    return p;
  };
  const size_t big = (size_t)MTOK * DD * sizeof(__bf16);   // 32 MB
  __bf16* xb   = (__bf16*)alloc(big);
  __bf16* wt[9];
  for (int i = 0; i < 9; ++i) wt[i] = (__bf16*)alloc((size_t)DD * DD * sizeof(__bf16));
  __bf16* Qb   = (__bf16*)alloc(big);
  __bf16* Kb   = (__bf16*)alloc(big);
  __bf16* Vb   = (__bf16*)alloc(big);
  __bf16* ctxb = (__bf16*)alloc(big);
  const size_t sm = (size_t)64 * DD * sizeof(__bf16);
  __bf16* mpb     = (__bf16*)alloc(sm);
  __bf16* summ_in = (__bf16*)alloc(sm);
  __bf16* sqb     = (__bf16*)alloc(sm);
  __bf16* skb     = (__bf16*)alloc(sm);
  __bf16* svb     = (__bf16*)alloc(sm);
  __bf16* sctx    = (__bf16*)alloc(sm);
  float*  summ_out = (float*)alloc((size_t)64 * DD * sizeof(float));

  // 0) precision staging
  cvt_bf16<<<(MTOK * DD + 255) / 256, 256, 0, stream>>>(x, xb, MTOK * DD);
  for (int i = 0; i < 9; ++i)
    cvt_t_bf16<<<(DD * DD) / 256, 256, 0, stream>>>(w_f32[i], wt[i]);

  // 1) local Q/K/V projections
  dim3 gBig(8, MTOK / 128), blk(256);
  gemm_wmma<<<gBig, blk, 0, stream>>>(xb, wt[0], lq_b, Qb, nullptr, nullptr, MTOK);
  gemm_wmma<<<gBig, blk, 0, stream>>>(xb, wt[1], lk_b, Kb, nullptr, nullptr, MTOK);
  gemm_wmma<<<gBig, blk, 0, stream>>>(xb, wt[2], lv_b, Vb, nullptr, nullptr, MTOK);

  // 2) local block attention with position bias
  local_attn<<<dim3(NWG, HH), blk, 0, stream>>>(Qb, Kb, Vb, pos_bias, ctxb);

  // 3) summary path
  meanpool<<<NWG, blk, 0, stream>>>(x, mpb);
  dim3 gSm(8, 1);
  gemm_wmma<<<gSm, blk, 0, stream>>>(mpb, wt[8], sum_b, summ_in, nullptr, nullptr, 64);
  gemm_wmma<<<gSm, blk, 0, stream>>>(summ_in, wt[4], sq_b, sqb, nullptr, nullptr, 64);
  gemm_wmma<<<gSm, blk, 0, stream>>>(summ_in, wt[5], sk_b, skb, nullptr, nullptr, 64);
  gemm_wmma<<<gSm, blk, 0, stream>>>(summ_in, wt[6], sv_b, svb, nullptr, nullptr, 64);
  summ_attn<<<dim3(4, HH), blk, 0, stream>>>(sqb, skb, svb, sctx);
  gemm_wmma<<<gSm, blk, 0, stream>>>(sctx, wt[7], so_b, nullptr, summ_out, nullptr, 64);

  // 4) output projection fused with block-summary broadcast add
  gemm_wmma<<<gBig, blk, 0, stream>>>(ctxb, wt[3], lo_b, nullptr, (float*)d_out,
                                      summ_out, MTOK);
}
